// GAT_Unet_71442486001971
// MI455X (gfx1250) — matrix operand order repaired and checked by reading, not compiled, not run on hardware
//
#include <hip/hip_runtime.h>
#include <hip/hip_bf16.h>

#define N_NODES 50000
#define N_EDGES 800000

typedef __attribute__((ext_vector_type(16))) __bf16 v16bf;
typedef __attribute__((ext_vector_type(8)))  __bf16 v8bf;
typedef __attribute__((ext_vector_type(8)))  float  v8f;

// ---------------------------------------------------------------------------
// WMMA GEMM: C[M,DOUT] = A[M,K] @ W[K,DOUT], A bf16 row-major, W supplied
// TRANSPOSED as Bt[DOUT,K] bf16 so both fragments load as contiguous b128s.
// One wave computes a 16-row x DOUT strip (NT = DOUT/16 tiles), 8 waves/block.
// Fragment layouts per CDNA5 ISA 7.12.2 (wave32), half = lane>>4, r = lane&15:
//   A 16x32: lane row = r; VGPR0-3 = K[8h .. 8h+7], VGPR4-7 = K[16+8h .. +7]
//     -> two 16B loads from row-major bf16 A
//   B 32x16: lane col = r; VGPR0-3 = K[8h .. 8h+7], VGPR4-7 = K[16+8h .. +7]
//     -> two 16B loads from Bt[col]
//   C/D 16x16 f32: element v -> row 8h+v, col r
// ---------------------------------------------------------------------------
template <int K, int DOUT>
__global__ __launch_bounds__(256)
void gemm_wmma(const __bf16* __restrict__ A, const __bf16* __restrict__ Bt,
               float* __restrict__ C, int M) {
  constexpr int NT = DOUT / 16;
  const int lane  = threadIdx.x & 31;
  const int wave  = threadIdx.x >> 5;
  const int mTile = blockIdx.x * 8 + wave;
  if (mTile >= M / 16) return;             // wave-uniform branch
  const int half = lane >> 4;
  const int r    = lane & 15;

  v8f acc[NT];
#pragma unroll
  for (int n = 0; n < NT; ++n) acc[n] = (v8f){};

  const __bf16* arow = A + (size_t)(mTile * 16 + r) * K;
#pragma unroll
  for (int k0 = 0; k0 < K; k0 += 32) {
    const v8bf alo = *(const v8bf*)(arow + k0 + half * 8);
    const v8bf ahi = *(const v8bf*)(arow + k0 + 16 + half * 8);
    const v16bf a = __builtin_shufflevector(alo, ahi, 0, 1, 2, 3, 4, 5, 6, 7,
                                            8, 9, 10, 11, 12, 13, 14, 15);
#pragma unroll
    for (int n = 0; n < NT; ++n) {
      const __bf16* bcol = Bt + (size_t)(n * 16 + r) * K + k0 + half * 8;
      const v8bf blo = *(const v8bf*)(bcol);
      const v8bf bhi = *(const v8bf*)(bcol + 16);
      const v16bf b = __builtin_shufflevector(blo, bhi, 0, 1, 2, 3, 4, 5, 6, 7,
                                              8, 9, 10, 11, 12, 13, 14, 15);
      acc[n] = __builtin_amdgcn_wmma_f32_16x16x32_bf16(
          /*neg_a=*/false, a, /*neg_b=*/false, b,
          /*c_mod=*/(short)0, acc[n], /*reuse_a=*/false, /*reuse_b=*/false);
    }
  }
#pragma unroll
  for (int n = 0; n < NT; ++n)
#pragma unroll
    for (int v = 0; v < 8; ++v)
      C[(size_t)(mTile * 16 + half * 8 + v) * DOUT + n * 16 + r] = acc[n][v];
}

// ---------------------------------------------------------------------------
// Conversions / staging
// ---------------------------------------------------------------------------
__global__ void f32_to_bf16(const float* __restrict__ in, __bf16* __restrict__ out,
                            int n) {
  int i = blockIdx.x * blockDim.x + threadIdx.x;
  if (i < n) out[i] = (__bf16)in[i];
}

// Wt[n,K] = (bf16) W[k,Nc]  (transpose + convert; <=24K elements, negligible)
__global__ void wt_bf16(const float* __restrict__ W, __bf16* __restrict__ Wt,
                        int K, int Nc) {
  int i = blockIdx.x * blockDim.x + threadIdx.x;
  if (i >= K * Nc) return;
  int k = i / Nc, n = i - k * Nc;
  Wt[(size_t)n * K + k] = (__bf16)W[i];
}

// ---------------------------------------------------------------------------
// Per-node attention projections: e_src[n] = h[n].a_s, e_dst[n] = h[n].a_d
// ---------------------------------------------------------------------------
__global__ void compute_e(const float* __restrict__ h, const float* __restrict__ a,
                          float* __restrict__ eS, float* __restrict__ eD,
                          int n, int d) {
  int i = blockIdx.x * blockDim.x + threadIdx.x;
  if (i >= n) return;
  const float* hr = h + (size_t)i * d;
  float s = 0.f, t = 0.f;
  for (int k = 0; k < d; ++k) {
    float hv = hr[k];
    s += hv * a[k];
    t += hv * a[d + k];
  }
  eS[i] = s;
  eD[i] = t;
}

// Monotonic uint ordering for f32 (atomicMax-based segment max).
__device__ __forceinline__ unsigned orderKey(float f) {
  unsigned u = __float_as_uint(f);
  return (u & 0x80000000u) ? ~u : (u | 0x80000000u);
}
__device__ __forceinline__ float keyToFloat(unsigned k) {
  unsigned u = (k & 0x80000000u) ? (k & 0x7FFFFFFFu) : ~k;
  return __uint_as_float(u);
}
#define NEG_INF_KEY 0x007FFFFFu  // orderKey(-inf)

__global__ void fill_f32(float* p, float v, int n) {
  int i = blockIdx.x * blockDim.x + threadIdx.x;
  if (i < n) p[i] = v;
}
__global__ void fill_u32(unsigned* p, unsigned v, int n) {
  int i = blockIdx.x * blockDim.x + threadIdx.x;
  if (i < n) p[i] = v;
}

// logit = leaky_relu(e_src[src] + e_dst[dst]); segment max via atomicMax(u32)
__global__ void edge_logit_max(const float* __restrict__ eS, const float* __restrict__ eD,
                               const int* __restrict__ src, const int* __restrict__ dst,
                               float* __restrict__ wE, unsigned* __restrict__ mKey,
                               int e) {
  int i = blockIdx.x * blockDim.x + threadIdx.x;
  if (i >= e) return;
  float l = eS[src[i]] + eD[dst[i]];
  l = (l > 0.f) ? l : 0.2f * l;
  wE[i] = l;
  atomicMax(&mKey[dst[i]], orderKey(l));
}

// w = exp(logit - m[dst]); segment sum via atomicAdd; overwrite logit with w
__global__ void edge_expsum(float* __restrict__ wE, const unsigned* __restrict__ mKey,
                            const int* __restrict__ dst, float* __restrict__ sSum,
                            int e) {
  int i = blockIdx.x * blockDim.x + threadIdx.x;
  if (i >= e) return;
  int dn = dst[i];
  float m = keyToFloat(mKey[dn]);
  if (!__builtin_isfinite(m)) m = 0.f;   // matches reference where() guard
  float w = __expf(wE[i] - m);
  wE[i] = w;
  atomicAdd(&sSum[dn], w);
}

// out[dst] += h[src] * (w / (s[dst]+1e-16)); one thread = one edge x 4 feats.
// L2-resident gather (float4) + same-cacheline f32 atomics.
__global__ void gat_aggregate(const float* __restrict__ h, const float* __restrict__ wE,
                              const float* __restrict__ sSum,
                              const int* __restrict__ src, const int* __restrict__ dst,
                              float* __restrict__ out, int e, int d) {
  const int chunks = d >> 2;
  int tid = blockIdx.x * blockDim.x + threadIdx.x;
  if (tid >= e * chunks) return;
  int ei = tid / chunks;
  int c  = (tid - ei * chunks) << 2;
  int sn = src[ei], dn = dst[ei];
  float alpha = wE[ei] / (sSum[dn] + 1e-16f);
  const float4 hv = *(const float4*)(h + (size_t)sn * d + c);
  float* o = out + (size_t)dn * d + c;
  atomicAdd(o + 0, hv.x * alpha);
  atomicAdd(o + 1, hv.y * alpha);
  atomicAdd(o + 2, hv.z * alpha);
  atomicAdd(o + 3, hv.w * alpha);
}

// ELU computed in f32, emitted as bf16 (all consumers are GEMMs)
__global__ void elu_bf16(const float* __restrict__ in, __bf16* __restrict__ out,
                         int n) {
  int i = blockIdx.x * blockDim.x + threadIdx.x;
  if (i >= n) return;
  float v = in[i];
  out[i] = (__bf16)((v > 0.f) ? v : (__expf(v) - 1.f));
}

// dst[n, da+db] = concat(A[n,da], B[n,db]) in bf16 (GEMM staging)
__global__ void concat_bf16(const __bf16* __restrict__ A, int da,
                            const __bf16* __restrict__ B, int db,
                            __bf16* __restrict__ dstp, int n) {
  int d = da + db;
  int i = blockIdx.x * blockDim.x + threadIdx.x;
  if (i >= n * d) return;
  int row = i / d, col = i - row * d;
  dstp[i] = (col < da) ? A[(size_t)row * da + col]
                       : B[(size_t)row * db + (col - da)];
}

// ---------------------------------------------------------------------------
extern "C" void kernel_launch(void* const* d_in, const int* in_sizes, int n_in,
                              void* d_out, int out_size, void* d_ws, size_t ws_size,
                              hipStream_t stream) {
  (void)in_sizes; (void)n_in; (void)out_size; (void)ws_size;
  const int Nn = N_NODES, Ee = N_EDGES;

  const float* x   = (const float*)d_in[0];
  const int*   src = (const int*)d_in[1];
  const int*   dst = (const int*)d_in[2];
  const float* W1  = (const float*)d_in[3];
  const float* W2  = (const float*)d_in[4];
  const float* W3  = (const float*)d_in[5];
  const float* W4  = (const float*)d_in[6];
  const float* a1  = (const float*)d_in[7];
  const float* a2  = (const float*)d_in[8];
  const float* a3  = (const float*)d_in[9];
  const float* a4  = (const float*)d_in[10];
  float* out = (float*)d_out;

  // ---- workspace carve-out: f32 region first (16B-aligned throughout) ----
  float* p = (float*)d_ws;
  float* bufH   = p; p += (size_t)Nn * 128;  // current-layer h (f32)
  float* bufAgg = p; p += (size_t)Nn * 128;  // pre-activation aggregate
  float* eS     = p; p += Nn;
  float* eD     = p; p += Nn;
  unsigned* mKey = (unsigned*)p; p += Nn;
  float* sSum   = p; p += Nn;
  float* wE     = p; p += Ee;                // logit, then softmax numerator
  // ---- bf16 region (GEMM staging) ----
  __bf16* q = (__bf16*)p;
  __bf16* xbf   = q; q += (size_t)Nn * 64;
  __bf16* d1bf  = q; q += (size_t)Nn * 64;
  __bf16* d2bf  = q; q += (size_t)Nn * 128;
  __bf16* u1bf  = q; q += (size_t)Nn * 64;
  __bf16* catbf = q; q += (size_t)Nn * 192;
  __bf16* W1t = q; q += 64 * 64;    // [64 ,64 ]
  __bf16* W2t = q; q += 128 * 64;   // [128,64 ]
  __bf16* W3t = q; q += 64 * 192;   // [64 ,192]
  __bf16* W4t = q; q += 64 * 128;   // [64 ,128]

  const int TB = 256;
  auto blocks = [](long long n, int tb) { return (int)((n + tb - 1) / tb); };
  const int gemmBlocks = blocks(Nn / 16, 8);  // 8 waves per block

  // edge-softmax + aggregation phase (shared by all 4 layers)
  auto edgePhase = [&](int dout, const float* a, float* aggOut) {
    compute_e<<<blocks(Nn, TB), TB, 0, stream>>>(bufH, a, eS, eD, Nn, dout);
    fill_u32<<<blocks(Nn, TB), TB, 0, stream>>>(mKey, NEG_INF_KEY, Nn);
    fill_f32<<<blocks(Nn, TB), TB, 0, stream>>>(sSum, 0.f, Nn);
    edge_logit_max<<<blocks(Ee, TB), TB, 0, stream>>>(eS, eD, src, dst, wE, mKey, Ee);
    edge_expsum<<<blocks(Ee, TB), TB, 0, stream>>>(wE, mKey, dst, sSum, Ee);
    fill_f32<<<blocks((long long)Nn * dout, TB), TB, 0, stream>>>(aggOut, 0.f, Nn * dout);
    gat_aggregate<<<blocks((long long)Ee * (dout / 4), TB), TB, 0, stream>>>(
        bufH, wE, sSum, src, dst, aggOut, Ee, dout);
  };

  // one-time staging: x -> bf16, weights -> transposed bf16
  f32_to_bf16<<<blocks(Nn * 64, TB), TB, 0, stream>>>(x, xbf, Nn * 64);
  wt_bf16<<<blocks(64 * 64, TB), TB, 0, stream>>>(W1, W1t, 64, 64);
  wt_bf16<<<blocks(64 * 128, TB), TB, 0, stream>>>(W2, W2t, 64, 128);
  wt_bf16<<<blocks(192 * 64, TB), TB, 0, stream>>>(W3, W3t, 192, 64);
  wt_bf16<<<blocks(128 * 64, TB), TB, 0, stream>>>(W4, W4t, 128, 64);

  // ---- encoder ----
  gemm_wmma<64, 64><<<gemmBlocks, TB, 0, stream>>>(xbf, W1t, bufH, Nn);
  edgePhase(64, a1, bufAgg);
  elu_bf16<<<blocks(Nn * 64, TB), TB, 0, stream>>>(bufAgg, d1bf, Nn * 64);

  gemm_wmma<64, 128><<<gemmBlocks, TB, 0, stream>>>(d1bf, W2t, bufH, Nn);
  edgePhase(128, a2, bufAgg);
  elu_bf16<<<blocks(Nn * 128, TB), TB, 0, stream>>>(bufAgg, d2bf, Nn * 128);

  // ---- decoder with U-Net skips ----
  concat_bf16<<<blocks(Nn * 192, TB), TB, 0, stream>>>(d2bf, 128, d1bf, 64, catbf, Nn);
  gemm_wmma<192, 64><<<gemmBlocks, TB, 0, stream>>>(catbf, W3t, bufH, Nn);
  edgePhase(64, a3, bufAgg);
  elu_bf16<<<blocks(Nn * 64, TB), TB, 0, stream>>>(bufAgg, u1bf, Nn * 64);

  concat_bf16<<<blocks(Nn * 128, TB), TB, 0, stream>>>(u1bf, 64, xbf, 64, catbf, Nn);
  gemm_wmma<128, 64><<<gemmBlocks, TB, 0, stream>>>(catbf, W4t, bufH, Nn);
  edgePhase(64, a4, out);  // final layer: no ELU, straight to d_out
}
